// PREF_65541200937504
// MI455X (gfx1250) — matrix operand order repaired and checked by reading, not compiled, not run on hardware
//
#include <hip/hip_runtime.h>
#include <math.h>

// ---------------------------------------------------------------------------
// Triplane-Fourier field evaluation for MI455X (gfx1250).
//
// Stage A: ifft2 of the three (16,8,160,160)-complex tensors as DFT matmuls
//          Y = X * G (G = 160x160 symmetric IDFT matrix, 1/160 per axis),
//          complex arithmetic as 4x v_wmma_f32_16x16x32_bf16 per K-step,
//          one wave per 16x16 output tile, K=160 in 5 steps of 32.
// Stage B: per-point bilinear gather of 128 complex channels per plane from
//          texel-major tables resident in the 192MB L2, folded with the
//          8-frequency cos/sin series. 8 lanes per point (lane = d), cross-
//          lane butterfly reduction with __shfl_xor.
// ---------------------------------------------------------------------------

#define LL      160
#define PLANE   (LL * LL)          // 25600
#define NCH     128                // 16 k * 8 d
#define CHPLANE (NCH * PLANE)      // 3,276,800
#define TEXF    (2 * NCH)          // 256 floats per texel (re,im interleaved)

typedef __attribute__((ext_vector_type(16))) __bf16 v16bf;
typedef __attribute__((ext_vector_type(8)))  float  v8f;

__device__ __forceinline__ float softplus_b10_dev(float x) {
  float bx = 10.0f * x;
  if (bx > 1.0f) return x;
  return log1pf(expf(fminf(bx, 1.0f))) * 0.1f;
}

// -------------------------------- G table ----------------------------------
__global__ void tp_ginit(float* __restrict__ Gr, float* __restrict__ Gi) {
  int idx = blockIdx.x * blockDim.x + threadIdx.x;
  if (idx >= PLANE) return;
  int j = idx / LL, m = idx % LL;
  int t = (j * m) % LL;                       // exact integer phase reduction
  float ang = (float)t * (6.2831853071795864769f / 160.0f);
  float s, c;
  sincosf(ang, &s, &c);
  Gr[idx] = c * (1.0f / 160.0f);
  Gi[idx] = s * (1.0f / 160.0f);
}

// --------------------- gather to canonical C[ch][h][w] ---------------------
__global__ void tp_gather(const float* __restrict__ re_in,
                          const float* __restrict__ im_in,
                          const float* __restrict__ alpha_p,
                          float* __restrict__ Cr, float* __restrict__ Ci,
                          int sk, int sd, int sh, int sw) {
  int idx = blockIdx.x * blockDim.x + threadIdx.x;
  if (idx >= CHPLANE) return;
  int ch  = idx / PLANE;
  int rem = idx - ch * PLANE;
  int h   = rem / LL;
  int w   = rem - h * LL;
  int k = ch >> 3, d = ch & 7;
  int off = k * sk + d * sd + h * sh + w * sw;
  float a = softplus_b10_dev(alpha_p[0]);
  Cr[idx] = a * re_in[off];
  Ci[idx] = a * im_in[off];
}

// ------------------------- WMMA complex GEMM tile --------------------------
__device__ __forceinline__ void load8_bf16(v16bf& v, int e0, const float* p) {
  float4 a = *(const float4*)(p);
  float4 b = *(const float4*)(p + 4);
  v[e0 + 0] = (__bf16)a.x; v[e0 + 1] = (__bf16)a.y;
  v[e0 + 2] = (__bf16)a.z; v[e0 + 3] = (__bf16)a.w;
  v[e0 + 4] = (__bf16)b.x; v[e0 + 5] = (__bf16)b.y;
  v[e0 + 6] = (__bf16)b.z; v[e0 + 7] = (__bf16)b.w;
}

// One wave computes a 16x16 tile of (Ar+iAi)[r0:r0+16, :] * (Gr+iGi)[:, c0:c0+16].
// A/B operands use contiguous k = 16*hs + e in both (16-bit A 16x32; B 32x16).
__device__ __forceinline__ void cplx_gemm_tile(
    const float* __restrict__ Ar, const float* __restrict__ Ai,  // [160][160]
    const float* __restrict__ Gr, const float* __restrict__ Gi,  // symmetric
    int r0, int c0, v8f& yr, v8f& yi) {
  const int lane = threadIdx.x & 31;
  const int nl = lane & 15;
  const int hs = lane >> 4;
  const float* ar_row = Ar + (r0 + nl) * LL;
  const float* ai_row = Ai + (r0 + nl) * LL;
  const float* gr_row = Gr + (c0 + nl) * LL;   // column slice via symmetry
  const float* gi_row = Gi + (c0 + nl) * LL;
  for (int kb = 0; kb < 5; ++kb) {
    const int K0 = kb * 32;
    v16bf a_r, a_i, b_r, b_i, b_ineg;
    // A 16x32: lane<16 -> K {0..15}; lane>=16 -> K {16..31}
    load8_bf16(a_r, 0, ar_row + K0 + 16 * hs);
    load8_bf16(a_r, 8, ar_row + K0 + 16 * hs + 8);
    load8_bf16(a_i, 0, ai_row + K0 + 16 * hs);
    load8_bf16(a_i, 8, ai_row + K0 + 16 * hs + 8);
    // B 32x16: lane n holds col n&15, K = 16*hs + e
    const float* pr = gr_row + K0 + 16 * hs;
    const float* pi = gi_row + K0 + 16 * hs;
#pragma unroll
    for (int e = 0; e < 16; ++e) {
      float br = pr[e], bi = pi[e];
      b_r[e]    = (__bf16)br;
      b_i[e]    = (__bf16)bi;
      b_ineg[e] = (__bf16)(-bi);   // bf16 WMMA has no A/B negate modifier
    }
    // Yr = Ar*Gr + Ai*(-Gi);  Yi = Ar*Gi + Ai*Gr
    yr = __builtin_amdgcn_wmma_f32_16x16x32_bf16(false, a_r, false, b_r,    (short)0, yr, false, false);
    yr = __builtin_amdgcn_wmma_f32_16x16x32_bf16(false, a_i, false, b_ineg, (short)0, yr, false, false);
    yi = __builtin_amdgcn_wmma_f32_16x16x32_bf16(false, a_r, false, b_i,    (short)0, yi, false, false);
    yi = __builtin_amdgcn_wmma_f32_16x16x32_bf16(false, a_i, false, b_r,    (short)0, yi, false, false);
  }
}

// Pass 1: D[ch][wm][h] = sum_w C[ch][h][w] * G[w][wm]   (transposed store)
__global__ void __launch_bounds__(32) tp_ifft_pass1(
    const float* __restrict__ Cr, const float* __restrict__ Ci,
    const float* __restrict__ Gr, const float* __restrict__ Gi,
    float* __restrict__ Dr, float* __restrict__ Di) {
  const int tile = blockIdx.x;             // 0..99
  const int ch   = blockIdx.y;             // 0..127
  const int r0 = (tile / 10) * 16, c0 = (tile % 10) * 16;
  v8f yr = {0.f,0.f,0.f,0.f,0.f,0.f,0.f,0.f};
  v8f yi = {0.f,0.f,0.f,0.f,0.f,0.f,0.f,0.f};
  cplx_gemm_tile(Cr + (size_t)ch * PLANE, Ci + (size_t)ch * PLANE, Gr, Gi, r0, c0, yr, yi);
  // D element (M=h local, N=wm local): lane holds N=nl, M = v + 8*hs (8 consecutive h)
  const int lane = threadIdx.x & 31, nl = lane & 15, hs = lane >> 4;
  float* dr = Dr + (size_t)ch * PLANE + (size_t)(c0 + nl) * LL + r0 + 8 * hs;
  float* di = Di + (size_t)ch * PLANE + (size_t)(c0 + nl) * LL + r0 + 8 * hs;
#pragma unroll
  for (int v = 0; v < 8; ++v) { dr[v] = yr[v]; di[v] = yi[v]; }
}

// Pass 2: E[ch][wm][hm] = sum_h D[ch][wm][h] * G[h][hm]; scatter into texel-
// major table T[hm][wm][2*ch + {0,1}].
__global__ void __launch_bounds__(32) tp_ifft_pass2(
    const float* __restrict__ Dr, const float* __restrict__ Di,
    const float* __restrict__ Gr, const float* __restrict__ Gi,
    float* __restrict__ T) {
  const int tile = blockIdx.x;
  const int ch   = blockIdx.y;
  const int r0 = (tile / 10) * 16, c0 = (tile % 10) * 16;
  v8f yr = {0.f,0.f,0.f,0.f,0.f,0.f,0.f,0.f};
  v8f yi = {0.f,0.f,0.f,0.f,0.f,0.f,0.f,0.f};
  cplx_gemm_tile(Dr + (size_t)ch * PLANE, Di + (size_t)ch * PLANE, Gr, Gi, r0, c0, yr, yi);
  const int lane = threadIdx.x & 31, nl = lane & 15, hs = lane >> 4;
  // element (M = wm = r0+8*hs+v, N = hm = c0+nl)
  float* base = T + ((size_t)(c0 + nl) * LL + r0 + 8 * hs) * TEXF + 2 * ch;
#pragma unroll
  for (int v = 0; v < 8; ++v) {
    float2 o = make_float2(yr[v], yi[v]);
    *(float2*)(base + (size_t)v * TEXF) = o;
  }
}

// ------------------------------ Stage B ------------------------------------
// 8 lanes per point (lane = d). Accumulates sum_d w*(re*c_d - im*s_d) for the
// 4 bilinear corners of one plane into acc[k], k=0..15.
__device__ __forceinline__ void accum_plane(const float* __restrict__ T,
                                            float gx, float gy, float t,
                                            float freq, float scale, int l,
                                            float acc[16]) {
  float s, c;
  sincosf(3.14159265358979323846f * (t + 1.0f) * freq, &s, &c);  // 2*pi*(t+1)/2*f
  c *= scale; s *= scale;
  float ix = (gx + 1.0f) * 0.5f * (float)(LL - 1);
  float iy = (gy + 1.0f) * 0.5f * (float)(LL - 1);
  float ix0 = floorf(ix), iy0 = floorf(iy);
  float wx = ix - ix0, wy = iy - iy0;
#pragma unroll
  for (int cy = 0; cy < 2; ++cy)
#pragma unroll
    for (int cx = 0; cx < 2; ++cx) {
      float xc = ix0 + (float)cx, yc = iy0 + (float)cy;
      bool valid = (xc >= 0.f) && (xc <= (float)(LL - 1)) &&
                   (yc >= 0.f) && (yc <= (float)(LL - 1));
      float w = (cx ? wx : 1.f - wx) * (cy ? wy : 1.f - wy) * (valid ? 1.f : 0.f);
      int ii = (int)fminf(fmaxf(xc, 0.f), (float)(LL - 1));
      int jj = (int)fminf(fmaxf(yc, 0.f), (float)(LL - 1));
      const float* base = T + ((size_t)jj * LL + ii) * TEXF + 2 * l;
      float wc = w * c, ws = w * s;
#pragma unroll
      for (int k = 0; k < 16; ++k) {          // channel ch = k*8 + l
        float2 v = *(const float2*)(base + k * 16);
        acc[k] = fmaf(v.x, wc, acc[k]);
        acc[k] = fmaf(-v.y, ws, acc[k]);
      }
    }
}

__global__ void tp_points(const float* __restrict__ xyz, const float* __restrict__ bnd,
                          const float* __restrict__ Tu, const float* __restrict__ Tv,
                          const float* __restrict__ Tw,
                          float* __restrict__ out, int N) {
  const int lane = threadIdx.x & 31;
  const int wave = (blockIdx.x * blockDim.x + threadIdx.x) >> 5;
  const int g = lane >> 3;          // point-within-wave
  const int l = lane & 7;           // d index
  const int p = wave * 4 + g;
  const bool active = p < N;
  float invb = 1.0f / bnd[0];
  float xs = 0.f, ys = 0.f, zs = 0.f;
  if (active) {
    xs = xyz[3 * p + 0] * invb;
    ys = xyz[3 * p + 1] * invb;
    zs = xyz[3 * p + 2] * invb;
  }
  const float freq  = (l == 0) ? 0.0f : (float)(1 << (l - 1));  // 0,1,2,4,...,64
  const float scale = (l == 0) ? 1.0f : 2.0f;
  float acc[16];
#pragma unroll
  for (int k = 0; k < 16; ++k) acc[k] = 0.f;
  if (active) {
    accum_plane(Tu, zs, ys, xs, freq, scale, l, acc);  // u-plane: (gy=ys, gx=zs), series in xs
    accum_plane(Tv, zs, xs, ys, freq, scale, l, acc);  // v-plane: (gy=xs, gx=zs), series in ys
    accum_plane(Tw, xs, ys, zs, freq, scale, l, acc);  // w-plane: (gy=ys, gx=xs), series in zs
  }
  // reduce over d (8 lanes) -- butterfly; all lanes participate
#pragma unroll
  for (int k = 0; k < 16; ++k) {
    acc[k] += __shfl_xor(acc[k], 1);
    acc[k] += __shfl_xor(acc[k], 2);
    acc[k] += __shfl_xor(acc[k], 4);
  }
  if (active) {
    // lane l writes k = 2l, 2l+1 (static-index select to avoid scratch)
    float o0 = 0.f, o1 = 0.f;
#pragma unroll
    for (int k = 0; k < 16; ++k) {
      if (k == 2 * l)     o0 = acc[k];
      if (k == 2 * l + 1) o1 = acc[k];
    }
    float2 o = make_float2(o0, o1);
    *(float2*)(out + (size_t)p * 16 + 2 * l) = o;
  }
}

// ------------------------------ launcher -----------------------------------
extern "C" void kernel_launch(void* const* d_in, const int* in_sizes, int n_in,
                              void* d_out, int out_size, void* d_ws, size_t ws_size,
                              hipStream_t stream) {
  const float* xyz     = (const float*)d_in[0];
  const float* bnd     = (const float*)d_in[1];
  const float* alpha_p = (const float*)d_in[2];
  const float* Pu_re   = (const float*)d_in[3];
  const float* Pu_im   = (const float*)d_in[4];
  const float* Pv_re   = (const float*)d_in[5];
  const float* Pv_im   = (const float*)d_in[6];
  const float* Pw_re   = (const float*)d_in[7];
  const float* Pw_im   = (const float*)d_in[8];
  float* out = (float*)d_out;
  const int N = in_sizes[0] / 3;

  // workspace layout (floats), total 32,819,200 f = 131.3 MB
  float* Gr = (float*)d_ws;
  float* Gi = Gr + PLANE;
  float* Tu = Gi + PLANE;
  float* Tv = Tu + (size_t)PLANE * TEXF;
  float* Tw = Tv + (size_t)PLANE * TEXF;
  float* Cr = Tw + (size_t)PLANE * TEXF;     // reused per tensor
  float* Ci = Cr + CHPLANE;
  float* Dr = Ci + CHPLANE;
  float* Di = Dr + CHPLANE;
  (void)n_in; (void)out_size; (void)ws_size;

  tp_ginit<<<(PLANE + 255) / 256, 256, 0, stream>>>(Gr, Gi);

  // canonical C[ch][h][w] gather strides per tensor:
  //  u: Pu[k][d][y][z] -> (h=y,w=z): sk=8*PLANE, sd=PLANE,  sh=LL,   sw=1
  //  v: Pv[k][x][d][z] -> (h=x,w=z): sk=8*PLANE, sd=LL,     sh=8*LL, sw=1
  //  w: Pw[k][x][y][d] -> (h=y,w=x): sk=8*PLANE, sd=1,      sh=8,    sw=8*LL
  struct TSpec { const float* re; const float* im; int sk, sd, sh, sw; float* T; };
  TSpec specs[3] = {
    { Pu_re, Pu_im, 8 * PLANE, PLANE,  LL,     1,      Tu },
    { Pv_re, Pv_im, 8 * PLANE, LL,     8 * LL, 1,      Tv },
    { Pw_re, Pw_im, 8 * PLANE, 1,      8,      8 * LL, Tw },
  };
  dim3 gtiles(100, NCH);   // 10x10 tiles of 16x16, 128 channels
  for (int t = 0; t < 3; ++t) {
    tp_gather<<<(CHPLANE + 255) / 256, 256, 0, stream>>>(
        specs[t].re, specs[t].im, alpha_p, Cr, Ci,
        specs[t].sk, specs[t].sd, specs[t].sh, specs[t].sw);
    tp_ifft_pass1<<<gtiles, 32, 0, stream>>>(Cr, Ci, Gr, Gi, Dr, Di);
    tp_ifft_pass2<<<gtiles, 32, 0, stream>>>(Dr, Di, Gr, Gi, specs[t].T);
  }

  const int nblocks = (N + 31) / 32;  // 256 threads = 8 waves = 32 points/block
  tp_points<<<nblocks, 256, 0, stream>>>(xyz, bnd, Tu, Tv, Tw, out, N);
}
